// GVP_31877247271159
// MI455X (gfx1250) — matrix operand order repaired and checked by reading, not compiled, not run on hardware
//
#include <hip/hip_runtime.h>

// ---------------------------------------------------------------------------
// GVP-style fused layer for MI455X (gfx1250, wave32, WMMA bf16 16x16x32)
// ---------------------------------------------------------------------------
typedef __attribute__((ext_vector_type(16))) __bf16 v16bf;
typedef __attribute__((ext_vector_type(8)))  float  v8f;

#define B_TOTAL 100000
#define V_IN    128
#define V_OUT   128
#define F_IN    512
#define F_OUT   512
#define N_CP    16
#define DIM_H   128
#define K_H     (DIM_H + N_CP)        // 144
#define N1      (DIM_H + 2 * N_CP)    // 160 : [Wh | Wcp] fused output width
#define KU_PAD  160                   // K for Vu matmul padded 144 -> 160
#define S_DIM   (F_IN + K_H)          // 656
#define S_PAD   672                   // padded to 21 * 32
#define TILE_M  32                    // batch rows per workgroup

static __device__ __forceinline__ unsigned short f32_to_bf16(float f) {
    union { float f; unsigned int u; } v; v.f = f;
    unsigned int r = v.u + 0x7FFFu + ((v.u >> 16) & 1u);   // round-to-nearest-even
    return (unsigned short)(r >> 16);
}

union Frag16 { uint4 q[2]; v16bf v; };

static __device__ __forceinline__ v16bf load_lds_frag(const unsigned short* p_lo,
                                                      const unsigned short* p_hi) {
    Frag16 u;
    u.q[0] = *reinterpret_cast<const uint4*>(p_lo);   // K = kb+half*8 .. +7   -> VGPR 0..3
    u.q[1] = *reinterpret_cast<const uint4*>(p_hi);   // K = kb+16+half*8 ..   -> VGPR 4..7
    return u.v;
}
static __device__ __forceinline__ v16bf load_gbl_frag(const unsigned int* p) {
    Frag16 u;
    const uint4* q = reinterpret_cast<const uint4*>(p);
    u.q[0] = q[0]; u.q[1] = q[1];
    return u.v;
}

// ---------------------------------------------------------------------------
// Prep: convert f32 row-major weight [K][N] (optionally concat of two
// matrices along N) into bf16 packed in WMMA B-fragment lane order.
// Fragment (nt,kt): lane L -> n = nt*16 + (L&15), ks = kt*32 + (L>>4)*8,
// dword v holds bf16(K=ks+2v+(v>>2)*8) | bf16(K=ks+2v+(v>>2)*8+1)<<16. 256 dwords / fragment.
// ---------------------------------------------------------------------------
__global__ void prep_frag_kernel(const float* __restrict__ Wa, int Na,
                                 const float* __restrict__ Wb,
                                 int K, int N, int Kpad,
                                 unsigned int* __restrict__ dst) {
    int t = blockIdx.x * blockDim.x + threadIdx.x;
    int KT = Kpad >> 5;
    int total = (N >> 4) * KT * 256;
    if (t >= total) return;
    int frag = t >> 8, r = t & 255;
    int lane = r >> 3, v = r & 7;
    int nt = frag / KT, kt = frag - nt * KT;
    int n  = (nt << 4) + (lane & 15);
    int ks = (kt << 5) + ((lane >> 4) << 3) + (v << 1) + ((v >> 2) << 3);
    float f0 = 0.f, f1 = 0.f;
    if (ks < K)
        f0 = (n < Na) ? Wa[(size_t)ks * Na + n] : Wb[(size_t)ks * (N - Na) + (n - Na)];
    if (ks + 1 < K)
        f1 = (n < Na) ? Wa[(size_t)(ks + 1) * Na + n] : Wb[(size_t)(ks + 1) * (N - Na) + (n - Na)];
    dst[t] = ((unsigned int)f32_to_bf16(f1) << 16) | (unsigned int)f32_to_bf16(f0);
}

// ---------------------------------------------------------------------------
// Fused main kernel: 32 batch rows per WG, 8 waves.
// ---------------------------------------------------------------------------
// LDS carve (bytes):
#define OFF_AVEC 0                    // ushort [3][32][128]  = 24576
#define OFF_VH   24576                // float  [3][32][160]  = 61440
#define OFF_ACAT 86016                // ushort [3][32][160]  = 30720
#define OFF_S    116736               // ushort [32][672]     = 43008
#define OFF_VU   159744               // float  [3][32][128]  = 49152
#define OFF_FO   208896               // ushort [32][512]     = 32768
#define SMEM_BYTES 241664

__global__ __launch_bounds__(256) void gvp_fused_kernel(
    const float* __restrict__ feats, const float* __restrict__ vectors,
    const float* __restrict__ bfv,   const float* __restrict__ bgv,
    const unsigned int* __restrict__ w1frag, const unsigned int* __restrict__ wufrag,
    const unsigned int* __restrict__ wffrag, const unsigned int* __restrict__ wgfrag,
    float* __restrict__ out_feats, float* __restrict__ out_vec)
{
    extern __shared__ unsigned char smem[];
    unsigned short* sAvec = (unsigned short*)(smem + OFF_AVEC);
    float*          sVh   = (float*)         (smem + OFF_VH);
    unsigned short* sAcat = (unsigned short*)(smem + OFF_ACAT);
    unsigned short* sS    = (unsigned short*)(smem + OFF_S);
    float*          sVu   = (float*)         (smem + OFF_VU);
    unsigned short* sFo   = (unsigned short*)(smem + OFF_FO);

    const int tid  = threadIdx.x;
    const int lane = tid & 31, wid = tid >> 5;
    const int half = lane >> 4, lm = lane & 15;
    const int mb   = blockIdx.x * TILE_M;

    // ---- phase 0: stage inputs into LDS as bf16 ----
    for (int i = tid; i < TILE_M * V_IN; i += 256) {
        int m = i >> 7, v = i & 127;
        const float* p = vectors + ((size_t)(mb + m) * V_IN + v) * 3;
        sAvec[(0 * TILE_M + m) * V_IN + v] = f32_to_bf16(p[0]);
        sAvec[(1 * TILE_M + m) * V_IN + v] = f32_to_bf16(p[1]);
        sAvec[(2 * TILE_M + m) * V_IN + v] = f32_to_bf16(p[2]);
    }
    for (int i = tid; i < TILE_M * F_IN; i += 256) {
        int m = i >> 9, f = i & 511;
        sS[m * S_PAD + f] = f32_to_bf16(feats[(size_t)(mb + m) * F_IN + f]);
    }
    for (int i = tid; i < TILE_M * 16; i += 256) {           // s padding 656..671
        int m = i >> 4, j = i & 15;
        sS[m * S_PAD + S_DIM + j] = 0;
    }
    for (int i = tid; i < 3 * TILE_M * 16; i += 256) {       // Acat padding 144..159
        int c = i / (TILE_M * 16), rr = i % (TILE_M * 16);
        sAcat[(c * TILE_M + (rr >> 4)) * KU_PAD + K_H + (rr & 15)] = 0;
    }
    if (tid == 0 && (mb + TILE_M) < B_TOTAL) {               // next-tile prefetch
        __builtin_prefetch(feats   + (size_t)(mb + TILE_M) * F_IN, 0, 1);
        __builtin_prefetch(vectors + (size_t)(mb + TILE_M) * V_IN * 3, 0, 1);
    }
    __syncthreads();

    // ---- phase A: [vectors_c 32x128] x [Wh|Wcp] 128x160 -> sVh[3][32][160] ----
    for (int job = wid; job < 30; job += 8) {
        int c = job / 10, nt = job % 10;
        v8f acc0 = {0.f,0.f,0.f,0.f,0.f,0.f,0.f,0.f};
        v8f acc1 = acc0;
        const unsigned short* ar0 = sAvec + (c * TILE_M + lm)      * V_IN;
        const unsigned short* ar1 = sAvec + (c * TILE_M + 16 + lm) * V_IN;
        for (int kt = 0; kt < 4; ++kt) {
            int kb = kt << 5;
            v16bf b  = load_gbl_frag(w1frag + (((size_t)nt * 4 + kt) << 8) + (lane << 3));
            v16bf a0 = load_lds_frag(ar0 + kb + half * 8, ar0 + kb + 16 + half * 8);
            v16bf a1 = load_lds_frag(ar1 + kb + half * 8, ar1 + kb + 16 + half * 8);
            acc0 = __builtin_amdgcn_wmma_f32_16x16x32_bf16(false, a0, false, b, (short)0, acc0, false, false);
            acc1 = __builtin_amdgcn_wmma_f32_16x16x32_bf16(false, a1, false, b, (short)0, acc1, false, false);
        }
        float* d0 = sVh + (c * TILE_M + half * 8) * N1 + (nt << 4) + lm;
        for (int r = 0; r < 8; ++r) { d0[r * N1] = acc0[r]; d0[(16 + r) * N1] = acc1[r]; }
    }
    __syncthreads();

    // ---- phase B: cross products, clamped norms, build Vu A-matrix + s ----
    for (int i = tid; i < TILE_M * K_H; i += 256) {
        int m = i / K_H, k = i - m * K_H;
        float x, y, z;
        if (k < DIM_H) {
            x = sVh[(0 * TILE_M + m) * N1 + k];
            y = sVh[(1 * TILE_M + m) * N1 + k];
            z = sVh[(2 * TILE_M + m) * N1 + k];
        } else {
            int p = k - DIM_H;
            float ax = sVh[(0 * TILE_M + m) * N1 + DIM_H + p];
            float ay = sVh[(1 * TILE_M + m) * N1 + DIM_H + p];
            float az = sVh[(2 * TILE_M + m) * N1 + DIM_H + p];
            float bx = sVh[(0 * TILE_M + m) * N1 + DIM_H + N_CP + p];
            float by = sVh[(1 * TILE_M + m) * N1 + DIM_H + N_CP + p];
            float bz = sVh[(2 * TILE_M + m) * N1 + DIM_H + N_CP + p];
            x = ay * bz - az * by;
            y = az * bx - ax * bz;
            z = ax * by - ay * bx;
        }
        sAcat[(0 * TILE_M + m) * KU_PAD + k] = f32_to_bf16(x);
        sAcat[(1 * TILE_M + m) * KU_PAD + k] = f32_to_bf16(y);
        sAcat[(2 * TILE_M + m) * KU_PAD + k] = f32_to_bf16(z);
        float nrm = sqrtf(fmaxf(x * x + y * y + z * z, 1e-8f));
        sS[m * S_PAD + F_IN + k] = f32_to_bf16(nrm);
    }
    __syncthreads();

    // ---- phase C: Vu (24 jobs) + feats_out = silu(s@Wf + bf) (32 jobs) ----
    for (int job = wid; job < 56; job += 8) {
        if (job < 24) {
            int c = job >> 3, nt = job & 7;
            v8f acc0 = {0.f,0.f,0.f,0.f,0.f,0.f,0.f,0.f};
            v8f acc1 = acc0;
            const unsigned short* ar0 = sAcat + (c * TILE_M + lm)      * KU_PAD;
            const unsigned short* ar1 = sAcat + (c * TILE_M + 16 + lm) * KU_PAD;
            for (int kt = 0; kt < 5; ++kt) {
                int kb = kt << 5;
                v16bf b  = load_gbl_frag(wufrag + (((size_t)nt * 5 + kt) << 8) + (lane << 3));
                v16bf a0 = load_lds_frag(ar0 + kb + half * 8, ar0 + kb + 16 + half * 8);
                v16bf a1 = load_lds_frag(ar1 + kb + half * 8, ar1 + kb + 16 + half * 8);
                acc0 = __builtin_amdgcn_wmma_f32_16x16x32_bf16(false, a0, false, b, (short)0, acc0, false, false);
                acc1 = __builtin_amdgcn_wmma_f32_16x16x32_bf16(false, a1, false, b, (short)0, acc1, false, false);
            }
            float* d0 = sVu + (c * TILE_M + half * 8) * V_OUT + (nt << 4) + lm;
            for (int r = 0; r < 8; ++r) { d0[r * V_OUT] = acc0[r]; d0[(16 + r) * V_OUT] = acc1[r]; }
        } else {
            int nt = job - 24;
            v8f acc0 = {0.f,0.f,0.f,0.f,0.f,0.f,0.f,0.f};
            v8f acc1 = acc0;
            const unsigned short* ar0 = sS + lm * S_PAD;
            const unsigned short* ar1 = sS + (16 + lm) * S_PAD;
            for (int kt = 0; kt < 21; ++kt) {
                int kb = kt << 5;
                v16bf b  = load_gbl_frag(wffrag + (((size_t)nt * 21 + kt) << 8) + (lane << 3));
                v16bf a0 = load_lds_frag(ar0 + kb + half * 8, ar0 + kb + 16 + half * 8);
                v16bf a1 = load_lds_frag(ar1 + kb + half * 8, ar1 + kb + 16 + half * 8);
                acc0 = __builtin_amdgcn_wmma_f32_16x16x32_bf16(false, a0, false, b, (short)0, acc0, false, false);
                acc1 = __builtin_amdgcn_wmma_f32_16x16x32_bf16(false, a1, false, b, (short)0, acc1, false, false);
            }
            int n = (nt << 4) + lm;
            float bias = bfv[n];
            for (int r = 0; r < 8; ++r) {
                int m0 = half * 8 + r, m1 = 16 + m0;
                float x0 = acc0[r] + bias, x1 = acc1[r] + bias;
                float s0 = x0 / (1.f + __expf(-x0));
                float s1 = x1 / (1.f + __expf(-x1));
                out_feats[(size_t)(mb + m0) * F_OUT + n] = s0;
                out_feats[(size_t)(mb + m1) * F_OUT + n] = s1;
                sFo[m0 * F_OUT + n] = f32_to_bf16(s0);
                sFo[m1 * F_OUT + n] = f32_to_bf16(s1);
            }
        }
    }
    __syncthreads();

    // ---- phase D: gating = feats_out@Wg + bg; vectors_out = sigmoid(g) * Vu ----
    {
        int nt = wid;                               // exactly 8 N-tiles, 8 waves
        v8f acc0 = {0.f,0.f,0.f,0.f,0.f,0.f,0.f,0.f};
        v8f acc1 = acc0;
        const unsigned short* ar0 = sFo + lm * F_OUT;
        const unsigned short* ar1 = sFo + (16 + lm) * F_OUT;
        for (int kt = 0; kt < 16; ++kt) {
            int kb = kt << 5;
            v16bf b  = load_gbl_frag(wgfrag + (((size_t)nt * 16 + kt) << 8) + (lane << 3));
            v16bf a0 = load_lds_frag(ar0 + kb + half * 8, ar0 + kb + 16 + half * 8);
            v16bf a1 = load_lds_frag(ar1 + kb + half * 8, ar1 + kb + 16 + half * 8);
            acc0 = __builtin_amdgcn_wmma_f32_16x16x32_bf16(false, a0, false, b, (short)0, acc0, false, false);
            acc1 = __builtin_amdgcn_wmma_f32_16x16x32_bf16(false, a1, false, b, (short)0, acc1, false, false);
        }
        int n = (nt << 4) + lm;
        float bias = bgv[n];
        for (int r = 0; r < 8; ++r) {
            int m0 = half * 8 + r, m1 = 16 + m0;
            float g0 = 1.f / (1.f + __expf(-(acc0[r] + bias)));
            float g1 = 1.f / (1.f + __expf(-(acc1[r] + bias)));
            float* o0 = out_vec + ((size_t)(mb + m0) * V_OUT + n) * 3;
            float* o1 = out_vec + ((size_t)(mb + m1) * V_OUT + n) * 3;
            o0[0] = g0 * sVu[(0 * TILE_M + m0) * V_OUT + n];
            o0[1] = g0 * sVu[(1 * TILE_M + m0) * V_OUT + n];
            o0[2] = g0 * sVu[(2 * TILE_M + m0) * V_OUT + n];
            o1[0] = g1 * sVu[(0 * TILE_M + m1) * V_OUT + n];
            o1[1] = g1 * sVu[(1 * TILE_M + m1) * V_OUT + n];
            o1[2] = g1 * sVu[(2 * TILE_M + m1) * V_OUT + n];
        }
    }
}

// ---------------------------------------------------------------------------
extern "C" void kernel_launch(void* const* d_in, const int* in_sizes, int n_in,
                              void* d_out, int out_size, void* d_ws, size_t ws_size,
                              hipStream_t stream) {
    (void)in_sizes; (void)n_in; (void)out_size; (void)ws_size;
    const float* feats   = (const float*)d_in[0];
    const float* vectors = (const float*)d_in[1];
    const float* Wh      = (const float*)d_in[2];
    const float* Wcp     = (const float*)d_in[3];
    const float* Wu      = (const float*)d_in[4];
    const float* Wf      = (const float*)d_in[5];
    const float* bfv     = (const float*)d_in[6];
    const float* Wg      = (const float*)d_in[7];
    const float* bgv     = (const float*)d_in[8];

    unsigned int* ws     = (unsigned int*)d_ws;
    unsigned int* w1frag = ws;                 // 10*4*256  = 10240 dw
    unsigned int* wufrag = ws + 10240;         // 8*5*256   = 10240 dw
    unsigned int* wffrag = ws + 20480;         // 32*21*256 = 172032 dw
    unsigned int* wgfrag = ws + 192512;        // 8*16*256  = 32768 dw  (total ~880 KB)

    auto nblk = [](int total) { return (total + 255) / 256; };
    prep_frag_kernel<<<nblk(10240),  256, 0, stream>>>(Wh, 128, Wcp,    128, 160, 128, w1frag);
    prep_frag_kernel<<<nblk(10240),  256, 0, stream>>>(Wu, 128, nullptr, 144, 128, 160, wufrag);
    prep_frag_kernel<<<nblk(172032), 256, 0, stream>>>(Wf, 512, nullptr, 656, 512, 672, wffrag);
    prep_frag_kernel<<<nblk(32768),  256, 0, stream>>>(Wg, 128, nullptr, 512, 128, 512, wgfrag);

    static int smem_cfg = [] {
        return (int)hipFuncSetAttribute((const void*)gvp_fused_kernel,
                                        hipFuncAttributeMaxDynamicSharedMemorySize,
                                        SMEM_BYTES);
    }();
    (void)smem_cfg;

    float* out_feats = (float*)d_out;
    float* out_vec   = out_feats + (size_t)B_TOTAL * F_OUT;
    gvp_fused_kernel<<<B_TOTAL / TILE_M, 256, SMEM_BYTES, stream>>>(
        feats, vectors, bfv, bgv, w1frag, wufrag, wffrag, wgfrag, out_feats, out_vec);
}
